// Block_7722351198258
// MI455X (gfx1250) — matrix operand order repaired and checked by reading, not compiled, not run on hardware
//
#include <hip/hip_runtime.h>
#include <hip/hip_bf16.h>
#include <math.h>

// ---------------------------------------------------------------------------
// MI455X (gfx1250, CDNA5, wave32) implementation of the audio/video cross-
// attention adapter block.  All GEMM-like work goes through
// V_WMMA_F32_16X16X32_F16 (f16 in, f32 accumulate), software-pipelined so
// WMMAs overlap the L2 load latency instead of s_wait_loadcnt 0 every step.
// ---------------------------------------------------------------------------

typedef __attribute__((ext_vector_type(16))) _Float16 v16h;
typedef __attribute__((ext_vector_type(8)))  float    v8f;

#define D_DIM    768
#define NHEAD    12
#define HDIM     64
#define NPATCH   196
#define BATCH    4
#define TFRM     8            // T == TS == 8
#define NTOK     (NPATCH*TFRM)        // 1568
#define MROWS    (BATCH*NTOK)         // 6272 (patch tokens, both streams)
#define MROWS_T  ((NPATCH+1)*32)      // 6304 (t stream incl. CLS)
#define DOWN_DIM 192
#define LNEPS    1e-5f
#define QSCALE   0.125f               // 64^-0.5

// ---------------------------------------------------------------------------
// WMMA helpers (contiguous k order k = 16*(lane/16) + elem, wave32)
// ---------------------------------------------------------------------------
__device__ __forceinline__ v8f wmma32(v16h a, v16h b, v8f c) {
  return __builtin_amdgcn_wmma_f32_16x16x32_f16(
      /*neg_a=*/false, a, /*neg_b=*/false, b,
      /*c_mod=*/(short)0, c, /*reuse_a=*/false, /*reuse_b=*/false);
}

// A tile 16(M)x32(K) from row-major f16, leading dim ld.
// lane<16: row M=lane,    elems [0..15]=K0..15
// lane>=16: row M=lane-16, elems [0..15]=K16..31
__device__ __forceinline__ v16h load_a_tile(const _Float16* __restrict__ p, int ld) {
  int lane = threadIdx.x & 31;
  int m  = lane & 15;
  int kb = (lane >> 4) << 4;                 // 0 or 16
  const _Float16* q = p + (size_t)m * ld + kb;
  v16h r;
#pragma unroll
  for (int i = 0; i < 16; ++i) r[i] = q[i];
  return r;
}

// B tile 32(K)x16(N) where B[k][n] = p[n*ld + k]  (i.e. W stored [N,K] row-major)
// lane<16: col N=lane, elems = K0..15 ; lane>=16: col N=lane-16, elems = K16..31
__device__ __forceinline__ v16h load_b_nk(const _Float16* __restrict__ p, int ld) {
  int lane = threadIdx.x & 31;
  int n  = lane & 15;
  int kb = (lane >> 4) << 4;                 // 0 or 16
  const _Float16* q = p + (size_t)n * ld + kb;
  v16h r;
#pragma unroll
  for (int i = 0; i < 16; ++i) r[i] = q[i];
  return r;
}

// B tile 32(K)x16(N) where B[k][n] = p[k*ld + n]  (K-major, used for P@V)
__device__ __forceinline__ v16h load_b_kn(const _Float16* __restrict__ p, int ld) {
  int lane = threadIdx.x & 31;
  int n  = lane & 15;
  int kb = (lane >> 4) << 4;
  v16h r;
#pragma unroll
  for (int i = 0; i < 16; ++i) r[i] = p[(size_t)(kb + i) * ld + n];
  return r;
}

// ---------------------------------------------------------------------------
// Elementwise / setup kernels
// ---------------------------------------------------------------------------
__global__ void cvt_f32_to_f16(const float* __restrict__ s, _Float16* __restrict__ d, int n) {
  int i = blockIdx.x * blockDim.x + threadIdx.x;
  if (i < n) d[i] = (_Float16)s[i];
}

// GEMM row g = b*1568 + n*8 + t  ->  token-major row n*32 + b*8 + t + base
__global__ void build_rowmap(int* __restrict__ mp, int base) {
  int g = blockIdx.x * blockDim.x + threadIdx.x;
  if (g < MROWS) {
    int b = g / NTOK, r = g % NTOK;
    int n = r / TFRM, t = r % TFRM;
    mp[g] = n * 32 + b * 8 + t + base;
  }
}

// Fused LayerNorm + positional add + f16 convert.
// Output row g = b*1568 + n*8 + t ; source row = (tokOffset+n)*32 + b*8 + t
__global__ __launch_bounds__(256)
void ln_pos_kernel(const float* __restrict__ src,
                   const float* __restrict__ gamma, const float* __restrict__ beta,
                   const float* __restrict__ space, const float* __restrict__ temp,
                   _Float16* __restrict__ dst, int tokOffset) {
  int g = blockIdx.x;
  int b = g / NTOK, r = g % NTOK;
  int n = r / TFRM, t = r % TFRM;
  int srow = (tokOffset + n) * 32 + b * 8 + t;
  const float* x = src + (size_t)srow * D_DIM;

  float v0[3], s = 0.f, sq = 0.f;
#pragma unroll
  for (int i = 0; i < 3; ++i) {
    float v = x[threadIdx.x + 256 * i];
    v0[i] = v; s += v; sq += v * v;
  }
#pragma unroll
  for (int off = 1; off < 32; off <<= 1) {
    s  += __shfl_xor(s,  off);
    sq += __shfl_xor(sq, off);
  }
  __shared__ float ss[8], ssq[8];
  if ((threadIdx.x & 31) == 0) { ss[threadIdx.x >> 5] = s; ssq[threadIdx.x >> 5] = sq; }
  __syncthreads();
  s = 0.f; sq = 0.f;
#pragma unroll
  for (int i = 0; i < 8; ++i) { s += ss[i]; sq += ssq[i]; }
  float mean = s * (1.f / D_DIM);
  float var  = sq * (1.f / D_DIM) - mean * mean;
  float rstd = rsqrtf(var + LNEPS);

  _Float16* d = dst + (size_t)g * D_DIM;
#pragma unroll
  for (int i = 0; i < 3; ++i) {
    int c = threadIdx.x + 256 * i;
    float y = (v0[i] - mean) * rstd * gamma[c] + beta[c]
              + space[(size_t)n * D_DIM + c] + temp[(size_t)t * D_DIM + c];
    d[c] = (_Float16)y;
  }
}

// LayerNorm of the 32 CLS rows of t_x -> rows [0,32) of X32/X16 (no pos add)
__global__ __launch_bounds__(256)
void cls_ln_kernel(const float* __restrict__ t_x,
                   const float* __restrict__ gamma, const float* __restrict__ beta,
                   float* __restrict__ x32, _Float16* __restrict__ x16) {
  int row = blockIdx.x;                       // 0..31 : CLS token, all (b,t)
  const float* x = t_x + (size_t)row * D_DIM;
  float v0[3], s = 0.f, sq = 0.f;
#pragma unroll
  for (int i = 0; i < 3; ++i) {
    float v = x[threadIdx.x + 256 * i];
    v0[i] = v; s += v; sq += v * v;
  }
#pragma unroll
  for (int off = 1; off < 32; off <<= 1) {
    s  += __shfl_xor(s,  off);
    sq += __shfl_xor(sq, off);
  }
  __shared__ float ss[8], ssq[8];
  if ((threadIdx.x & 31) == 0) { ss[threadIdx.x >> 5] = s; ssq[threadIdx.x >> 5] = sq; }
  __syncthreads();
  s = 0.f; sq = 0.f;
#pragma unroll
  for (int i = 0; i < 8; ++i) { s += ss[i]; sq += ssq[i]; }
  float mean = s * (1.f / D_DIM);
  float rstd = rsqrtf(sq * (1.f / D_DIM) - mean * mean + LNEPS);
#pragma unroll
  for (int i = 0; i < 3; ++i) {
    int c = threadIdx.x + 256 * i;
    float y = (v0[i] - mean) * rstd * gamma[c] + beta[c];
    x32[(size_t)row * D_DIM + c] = y;
    x16[(size_t)row * D_DIM + c] = (_Float16)y;
  }
}

// ---------------------------------------------------------------------------
// Generic WMMA GEMM:  out[m,n] = epi( (sum_k A[m,k]*W[n,k] + bias[n]) * alpha )
// Each wave computes a 32(M) x 64(N) tile (8 accumulators).  The k-loop is
// software pipelined: tiles for step k+32 are fetched while the 8 WMMAs of
// step k execute, so waits cover loads issued one iteration earlier.
// Optional: exact GELU, output row remap, f32 and/or f16 stores, 2 residuals.
// ---------------------------------------------------------------------------
__global__ __launch_bounds__(256)
void wmma_gemm_kernel(const _Float16* __restrict__ A, const _Float16* __restrict__ W,
                      const float* __restrict__ bias,
                      int M, int N, int K, float alpha, int gelu_flag,
                      const int* __restrict__ rowmap,
                      float* __restrict__ out32, _Float16* __restrict__ out16,
                      const float* __restrict__ res0, const float* __restrict__ res1) {
  int wave   = blockIdx.x * (blockDim.x >> 5) + (threadIdx.x >> 5);
  int tilesN = N >> 6;
  int tilesM = M >> 5;                        // 32-row tiles (M % 32 == 0)
  if (wave >= tilesM * tilesN) return;
  int tm = wave / tilesN, tn = wave % tilesN;

  const _Float16* Ap0 = A + (size_t)tm * 32 * K;
  const _Float16* Ap1 = Ap0 + (size_t)16 * K;
  const _Float16* Wp  = W + (size_t)tn * 64 * K;

  v8f acc[2][4] = {};

  // prologue: first k-block
  v16h a0 = load_a_tile(Ap0, K);
  v16h a1 = load_a_tile(Ap1, K);
  v16h b0 = load_b_nk(Wp + (size_t)0 * 16 * K, K);
  v16h b1 = load_b_nk(Wp + (size_t)1 * 16 * K, K);
  v16h b2 = load_b_nk(Wp + (size_t)2 * 16 * K, K);
  v16h b3 = load_b_nk(Wp + (size_t)3 * 16 * K, K);

  for (int k0 = 32; k0 < K; k0 += 32) {
    // prefetch next k-block before consuming current registers
    v16h a0n = load_a_tile(Ap0 + k0, K);
    v16h a1n = load_a_tile(Ap1 + k0, K);
    v16h b0n = load_b_nk(Wp + (size_t)0 * 16 * K + k0, K);
    v16h b1n = load_b_nk(Wp + (size_t)1 * 16 * K + k0, K);
    v16h b2n = load_b_nk(Wp + (size_t)2 * 16 * K + k0, K);
    v16h b3n = load_b_nk(Wp + (size_t)3 * 16 * K + k0, K);

    acc[0][0] = wmma32(a0, b0, acc[0][0]);
    acc[0][1] = wmma32(a0, b1, acc[0][1]);
    acc[0][2] = wmma32(a0, b2, acc[0][2]);
    acc[0][3] = wmma32(a0, b3, acc[0][3]);
    acc[1][0] = wmma32(a1, b0, acc[1][0]);
    acc[1][1] = wmma32(a1, b1, acc[1][1]);
    acc[1][2] = wmma32(a1, b2, acc[1][2]);
    acc[1][3] = wmma32(a1, b3, acc[1][3]);

    a0 = a0n; a1 = a1n; b0 = b0n; b1 = b1n; b2 = b2n; b3 = b3n;
  }
  // epilogue k-block
  acc[0][0] = wmma32(a0, b0, acc[0][0]);
  acc[0][1] = wmma32(a0, b1, acc[0][1]);
  acc[0][2] = wmma32(a0, b2, acc[0][2]);
  acc[0][3] = wmma32(a0, b3, acc[0][3]);
  acc[1][0] = wmma32(a1, b0, acc[1][0]);
  acc[1][1] = wmma32(a1, b1, acc[1][1]);
  acc[1][2] = wmma32(a1, b2, acc[1][2]);
  acc[1][3] = wmma32(a1, b3, acc[1][3]);

  int lane = threadIdx.x & 31;
  int n  = lane & 15;
  int mb = (lane >> 4) << 3;
#pragma unroll
  for (int h = 0; h < 2; ++h) {
#pragma unroll
    for (int c = 0; c < 4; ++c) {
#pragma unroll
      for (int r = 0; r < 8; ++r) {
        int col  = (tn << 6) + (c << 4) + n;
        int grow = (tm << 5) + (h << 4) + mb + r;
        float v = (acc[h][c][r] + bias[col]) * alpha;
        if (gelu_flag) v = 0.5f * v * (1.0f + erff(v * 0.70710678118654752f));
        int orow = rowmap ? rowmap[grow] : grow;
        size_t idx = (size_t)orow * N + col;
        if (res0) v += res0[idx];
        if (res1) v += res1[idx];
        if (out32) out32[idx] = v;
        if (out16) out16[idx] = (_Float16)v;
      }
    }
  }
}

// ---------------------------------------------------------------------------
// Flash-style cross attention.  Q: [B*Nq,768] f16 (pre-scaled), KV: [B*Nk,1536]
// (K cols 0..767, V cols 768..1535).  One wave per (b, h, 16-query tile).
// K-tiles for chunk k+32 are prefetched at the top of the loop, and V-tiles
// are fetched before the softmax so their latency hides under the VALU work.
// ---------------------------------------------------------------------------
__global__ __launch_bounds__(32)
void attn_kernel(const _Float16* __restrict__ Q, const _Float16* __restrict__ KV,
                 _Float16* __restrict__ O, int Nq, int Nk) {
  __shared__ _Float16 pbuf[16 * 32];
  const int ld2 = 2 * D_DIM;

  int qt = blockIdx.x;
  int bh = blockIdx.y;
  int b = bh / NHEAD, h = bh % NHEAD;
  int lane = threadIdx.x & 31;
  int n  = lane & 15;
  int mb = (lane >> 4) << 3;

  const _Float16* Qp = Q  + ((size_t)(b * Nq + qt * 16)) * D_DIM + h * HDIM;
  const _Float16* Kp = KV + (size_t)b * Nk * ld2 + h * HDIM;
  const _Float16* Vp = Kp + D_DIM;

  v16h aq0 = load_a_tile(Qp,      D_DIM);   // head dims 0..31
  v16h aq1 = load_a_tile(Qp + 32, D_DIM);   // head dims 32..63

  float mrow[8], lrow[8];
#pragma unroll
  for (int r = 0; r < 8; ++r) { mrow[r] = -1e30f; lrow[r] = 0.f; }
  v8f o[4] = {};

  // prologue: K tiles for keys 0..31
  v16h kA0 = load_b_nk(Kp, ld2);                         // keys 0-15,  d 0-31
  v16h kA1 = load_b_nk(Kp + 32, ld2);                    // keys 0-15,  d 32-63
  v16h kB0 = load_b_nk(Kp + (size_t)16 * ld2, ld2);      // keys 16-31, d 0-31
  v16h kB1 = load_b_nk(Kp + (size_t)16 * ld2 + 32, ld2); // keys 16-31, d 32-63

  for (int k0 = 0; k0 < Nk; k0 += 32) {
    // issue V loads for this chunk + K loads for the next chunk first
    const _Float16* Vb = Vp + (size_t)k0 * ld2;
    v16h bv0 = load_b_kn(Vb + 0,  ld2);
    v16h bv1 = load_b_kn(Vb + 16, ld2);
    v16h bv2 = load_b_kn(Vb + 32, ld2);
    v16h bv3 = load_b_kn(Vb + 48, ld2);

    int kn = (k0 + 32 < Nk) ? (k0 + 32) : k0;            // clamp (unused last iter)
    const _Float16* Kn0 = Kp + (size_t)kn * ld2;
    const _Float16* Kn1 = Kp + (size_t)(kn + 16) * ld2;
    v16h nA0 = load_b_nk(Kn0, ld2);
    v16h nA1 = load_b_nk(Kn0 + 32, ld2);
    v16h nB0 = load_b_nk(Kn1, ld2);
    v16h nB1 = load_b_nk(Kn1 + 32, ld2);

    // scores: two 16x16 key sub-tiles, K(head dim)=64 as 2 x WMMA each
    v8f s0 = {}, s1 = {};
    s0 = wmma32(aq0, kA0, s0);
    s0 = wmma32(aq1, kA1, s0);
    s1 = wmma32(aq0, kB0, s1);
    s1 = wmma32(aq1, kB1, s1);

    // online softmax (rows live in a fixed 16-lane half -> shfl_xor 1,2,4,8)
    float sc[8];
#pragma unroll
    for (int r = 0; r < 8; ++r) {
      float tmax = fmaxf(s0[r], s1[r]);
#pragma unroll
      for (int off = 1; off < 16; off <<= 1) tmax = fmaxf(tmax, __shfl_xor(tmax, off));
      float nm = fmaxf(mrow[r], tmax);
      float p0 = __expf(s0[r] - nm);
      float p1 = __expf(s1[r] - nm);
      float rs = p0 + p1;
#pragma unroll
      for (int off = 1; off < 16; off <<= 1) rs += __shfl_xor(rs, off);
      sc[r]   = __expf(mrow[r] - nm);
      lrow[r] = lrow[r] * sc[r] + rs;
      mrow[r] = nm;
      pbuf[(mb + r) * 32 + n]      = (_Float16)p0;
      pbuf[(mb + r) * 32 + 16 + n] = (_Float16)p1;
    }
#pragma unroll
    for (int c = 0; c < 4; ++c)
#pragma unroll
      for (int r = 0; r < 8; ++r) o[c][r] *= sc[r];

    __syncthreads();                         // LDS writes -> A-layout reads
    v16h ap = load_a_tile(pbuf, 32);         // P tile 16x32 in A layout
    o[0] = wmma32(ap, bv0, o[0]);
    o[1] = wmma32(ap, bv1, o[1]);
    o[2] = wmma32(ap, bv2, o[2]);
    o[3] = wmma32(ap, bv3, o[3]);
    __syncthreads();                         // reads done before next overwrite

    kA0 = nA0; kA1 = nA1; kB0 = nB0; kB1 = nB1;
  }

  // normalize + store f16 [B*Nq, 768] at head column block
#pragma unroll
  for (int r = 0; r < 8; ++r) {
    float inv = 1.0f / lrow[r];
    size_t row = (size_t)(b * Nq + qt * 16 + mb + r);
#pragma unroll
    for (int c = 0; c < 4; ++c)
      O[row * D_DIM + h * HDIM + c * 16 + n] = (_Float16)(o[c][r] * inv);
  }
}

// ---------------------------------------------------------------------------
// Host-side orchestration
// ---------------------------------------------------------------------------
extern "C" void kernel_launch(void* const* d_in, const int* in_sizes, int n_in,
                              void* d_out, int out_size, void* d_ws, size_t ws_size,
                              hipStream_t stream) {
  (void)in_sizes; (void)n_in; (void)out_size; (void)ws_size;

  const float* s_x    = (const float*)d_in[0];
  const float* t_x    = (const float*)d_in[1];
  const float* ln_s_g = (const float*)d_in[2];
  const float* ln_s_b = (const float*)d_in[3];
  const float* ln_t_g = (const float*)d_in[4];
  const float* ln_t_b = (const float*)d_in[5];

  // ---- workspace carve ----
  size_t off = 0;
  auto carve = [&](size_t bytes) -> char* {
    char* r = (char*)d_ws + off;
    off = (off + bytes + 255) & ~(size_t)255;
    return r;
  };
  _Float16 *qw16[2], *kvw16[2], *pw16[2];
  for (int s2 = 0; s2 < 2; ++s2) {
    qw16[s2]  = (_Float16*)carve((size_t)D_DIM * D_DIM * 2);
    kvw16[s2] = (_Float16*)carve((size_t)2 * D_DIM * D_DIM * 2);
    pw16[s2]  = (_Float16*)carve((size_t)D_DIM * D_DIM * 2);
  }
  _Float16* aSw1 = (_Float16*)carve((size_t)DOWN_DIM * D_DIM * 2);
  _Float16* aSw2 = (_Float16*)carve((size_t)D_DIM * DOWN_DIM * 2);
  _Float16* aTw1 = (_Float16*)carve((size_t)DOWN_DIM * D_DIM * 2);
  _Float16* aTw2 = (_Float16*)carve((size_t)D_DIM * DOWN_DIM * 2);

  int* map_t = (int*)carve((size_t)MROWS * 4);
  int* map_s = (int*)carve((size_t)MROWS * 4);

  _Float16* Aq   = (_Float16*)carve((size_t)MROWS * D_DIM * 2);
  _Float16* Akv  = (_Float16*)carve((size_t)MROWS * D_DIM * 2);
  _Float16* Qb   = (_Float16*)carve((size_t)MROWS * D_DIM * 2);
  _Float16* KVb  = (_Float16*)carve((size_t)MROWS * 2 * D_DIM * 2);
  _Float16* AttO = (_Float16*)carve((size_t)MROWS * D_DIM * 2);
  float*    X32  = (float*)   carve((size_t)MROWS_T * D_DIM * 4);
  _Float16* X16  = (_Float16*)carve((size_t)MROWS_T * D_DIM * 2);
  _Float16* Hd   = (_Float16*)carve((size_t)MROWS_T * DOWN_DIM * 2);

  // ---- weight fp32 -> f16 ----
  auto cvt = [&](int idx, _Float16* dst, int n) {
    cvt_f32_to_f16<<<(n + 255) / 256, 256, 0, stream>>>((const float*)d_in[idx], dst, n);
  };
  cvt(10, qw16[0], D_DIM * D_DIM);  cvt(12, kvw16[0], 2 * D_DIM * D_DIM);  cvt(14, pw16[0], D_DIM * D_DIM);
  cvt(20, qw16[1], D_DIM * D_DIM);  cvt(22, kvw16[1], 2 * D_DIM * D_DIM);  cvt(24, pw16[1], D_DIM * D_DIM);
  cvt(26, aSw1, DOWN_DIM * D_DIM);  cvt(28, aSw2, D_DIM * DOWN_DIM);
  cvt(30, aTw1, DOWN_DIM * D_DIM);  cvt(32, aTw2, D_DIM * DOWN_DIM);

  build_rowmap<<<(MROWS + 255) / 256, 256, 0, stream>>>(map_t, 32);  // +32: CLS rows first
  build_rowmap<<<(MROWS + 255) / 256, 256, 0, stream>>>(map_s, 0);

  float* s_out = (float*)d_out;                                  // [196*32,768]
  float* t_out = s_out + (size_t)NPATCH * 32 * D_DIM;            // [197*32,768]

  for (int side = 0; side < 2; ++side) {          // 0 = s2t, 1 = t2s
    bool s2t = (side == 0);
    int pbase = s2t ? 6 : 16;
    const float* clip_space = (const float*)d_in[pbase + 0];
    const float* vmae_space = (const float*)d_in[pbase + 1];
    const float* clip_temp  = (const float*)d_in[pbase + 2];
    const float* vmae_temp  = (const float*)d_in[pbase + 3];
    const float* qbias      = (const float*)d_in[pbase + 5];
    const float* kvbias     = (const float*)d_in[pbase + 7];
    const float* pbias      = (const float*)d_in[pbase + 9];

    // queries / kv inputs: fused LN + pos + f16
    if (s2t) {
      ln_pos_kernel<<<MROWS, 256, 0, stream>>>(t_x, ln_t_g, ln_t_b, vmae_space, vmae_temp, Aq, 1);
      ln_pos_kernel<<<MROWS, 256, 0, stream>>>(s_x, ln_s_g, ln_s_b, clip_space, clip_temp, Akv, 0);
    } else {
      ln_pos_kernel<<<MROWS, 256, 0, stream>>>(s_x, ln_s_g, ln_s_b, clip_space, clip_temp, Aq, 0);
      ln_pos_kernel<<<MROWS, 256, 0, stream>>>(t_x, ln_t_g, ln_t_b, vmae_space, vmae_temp, Akv, 1);
    }

    // Q projection (scale folded in), KV projection
    {
      int tiles = (MROWS / 32) * (D_DIM / 64);
      wmma_gemm_kernel<<<(tiles + 7) / 8, 256, 0, stream>>>(
          Aq, qw16[side], qbias, MROWS, D_DIM, D_DIM, QSCALE, 0,
          nullptr, nullptr, Qb, nullptr, nullptr);
    }
    {
      int tiles = (MROWS / 32) * (2 * D_DIM / 64);
      wmma_gemm_kernel<<<(tiles + 7) / 8, 256, 0, stream>>>(
          Akv, kvw16[side], kvbias, MROWS, 2 * D_DIM, D_DIM, 1.0f, 0,
          nullptr, nullptr, KVb, nullptr, nullptr);
    }

    // attention
    attn_kernel<<<dim3(NTOK / 16, BATCH * NHEAD), 32, 0, stream>>>(Qb, KVb, AttO, NTOK, NTOK);

    // output projection with token-major reorder -> X32/X16
    {
      int tiles = (MROWS / 32) * (D_DIM / 64);
      wmma_gemm_kernel<<<(tiles + 7) / 8, 256, 0, stream>>>(
          AttO, pw16[side], pbias, MROWS, D_DIM, D_DIM, 1.0f, 0,
          s2t ? map_t : map_s, X32, X16, nullptr, nullptr);
    }
    if (s2t)
      cls_ln_kernel<<<32, 256, 0, stream>>>(t_x, ln_t_g, ln_t_b, X32, X16);

    // adapter: GELU bottleneck + double residual into d_out
    int Mx = s2t ? MROWS_T : MROWS;
    const _Float16* w1 = s2t ? aTw1 : aSw1;
    const _Float16* w2 = s2t ? aTw2 : aSw2;
    const float* b1 = (const float*)d_in[s2t ? 31 : 27];
    const float* b2 = (const float*)d_in[s2t ? 33 : 29];
    {
      int tiles = (Mx / 32) * (DOWN_DIM / 64);
      wmma_gemm_kernel<<<(tiles + 7) / 8, 256, 0, stream>>>(
          X16, w1, b1, Mx, DOWN_DIM, D_DIM, 1.0f, /*gelu=*/1,
          nullptr, nullptr, Hd, nullptr, nullptr);
    }
    {
      int tiles = (Mx / 32) * (D_DIM / 64);
      const float* res0 = s2t ? t_x : s_x;
      float* outp = s2t ? t_out : s_out;
      wmma_gemm_kernel<<<(tiles + 7) / 8, 256, 0, stream>>>(
          Hd, w2, b2, Mx, D_DIM, DOWN_DIM, 1.0f, 0,
          nullptr, outp, nullptr, res0, X32);
    }
  }
}
